// QuantizedLayer_19576460935501
// MI455X (gfx1250) — matrix operand order repaired and checked by reading, not compiled, not run on hardware
//
#include <hip/hip_runtime.h>

typedef __attribute__((ext_vector_type(16))) _Float16 v16h;
typedef __attribute__((ext_vector_type(8)))  _Float16 v8h;
typedef __attribute__((ext_vector_type(4)))  _Float16 v4h;
typedef __attribute__((ext_vector_type(8)))  float    v8f;

#ifndef __has_builtin
#define __has_builtin(x) 0
#endif

// CDNA5 async global->LDS DMA path (ASYNCcnt). Guarded: falls back to
// global_load_b128 -> ds_store_b128 register staging if builtins are absent.
#if __has_builtin(__builtin_amdgcn_global_load_async_to_lds_b128)
#define HAVE_ASYNC_LDS 1
#else
#define HAVE_ASYNC_LDS 0
#endif

#define TILE 128
#define BK   32
#define LDT  40   // padded LDS row stride in halves (80B) -> conflict-free 16-row frag reads

__device__ __forceinline__ void wait_async0() {
#if __has_builtin(__builtin_amdgcn_s_wait_asynccnt)
  __builtin_amdgcn_s_wait_asynccnt(0);
#else
  asm volatile("s_wait_asynccnt 0x0" ::: "memory");
#endif
}

#if HAVE_ASYNC_LDS
// Builtin signature (from hipcc diagnostic): params are pointers to
// `int __attribute__((vector_size(16)))` in AS1 (global) / AS3 (LDS).
typedef int b128i __attribute__((vector_size(16)));
typedef __attribute__((address_space(1))) b128i* gas1_b128;
typedef __attribute__((address_space(3))) b128i* las3_b128;

// Copy 32B per lane: global g -> LDS l. ISA: INST_OFFSET is added to BOTH the
// global and LDS address, so one (g,l) pair serves both 16B halves.
__device__ __forceinline__ void async_cp32(const _Float16* g, _Float16* l) {
  auto gv = (__attribute__((address_space(1))) void*)const_cast<_Float16*>(g);
  auto lv = (__attribute__((address_space(3))) void*)l;
  __builtin_amdgcn_global_load_async_to_lds_b128((gas1_b128)gv, (las3_b128)lv, 0, 0);
  __builtin_amdgcn_global_load_async_to_lds_b128((gas1_b128)gv, (las3_b128)lv, 16, 0);
}
#endif

// ---------------- fp32 -> fp16 quantize (RNE, denorms preserved) ------------
__global__ __launch_bounds__(256) void qcvt_kernel(const float* __restrict__ in,
                                                   _Float16* __restrict__ out,
                                                   int n) {
  int i = (blockIdx.x * 256 + threadIdx.x) * 4;
  if (i + 3 < n) {
    float4 v = *(const float4*)(in + i);
    v4h h;
    h.x = (_Float16)v.x; h.y = (_Float16)v.y;
    h.z = (_Float16)v.z; h.w = (_Float16)v.w;
    *(v4h*)(out + i) = h;
  } else {
    for (; i < n; ++i) out[i] = (_Float16)in[i];
  }
}

// ---------------- WMMA GEMM: C = Aq * Bq^T + q(bias) ------------------------
// A: M x K (x), B: N x K (weight, used transposed) -> both K-contiguous (NT GEMM).
// PREQ=true : A/B are pre-quantized _Float16 in workspace (fit in 192MB L2);
//             staged into LDS via async DMA when available.
// PREQ=false: A/B are fp32; quantize in-register after global load.
template <bool PREQ>
__global__ __launch_bounds__(256) void gemm_wmma_kernel(
    const void* __restrict__ Ap, const void* __restrict__ Bp,
    const float* __restrict__ bias, float* __restrict__ C,
    int M, int N, int K) {
  __shared__ alignas(16) _Float16 sA[2][TILE * LDT];
  __shared__ alignas(16) _Float16 sB[2][TILE * LDT];

  const int t    = threadIdx.x;
  const int lane = t & 31;
  const int wave = t >> 5;
  const int hid  = lane >> 4;  // half-wave id (0/1)
  const int lr   = lane & 15;
  const int wm   = wave >> 2;  // 0..1 -> 64 rows each
  const int wn   = wave & 3;   // 0..3 -> 32 cols each

  const int bm = blockIdx.y * TILE;
  const int bn = blockIdx.x * TILE;

  // global->LDS staging: each thread owns one 16-half (32B) chunk per matrix per K-step
  const int ldRow = t >> 1;
  const int ldCol = (t & 1) * 16;
  const int ldsOff = ldRow * LDT + ldCol;

  const size_t aBase = (size_t)(bm + ldRow) * K + ldCol;
  const size_t bBase = (size_t)(bn + ldRow) * K + ldCol;

  v8f acc[4][2] = {};
  v8h a0, a1, b0, b1;

  auto fetch = [&](size_t off, const void* P, v8h& lo, v8h& hi) {
    if (PREQ) {
      const _Float16* p = (const _Float16*)P + off;
      lo = *(const v8h*)p;
      hi = *(const v8h*)(p + 8);
    } else {
      const float* p = (const float*)P + off;
      float4 f0 = *(const float4*)(p);
      float4 f1 = *(const float4*)(p + 4);
      float4 f2 = *(const float4*)(p + 8);
      float4 f3 = *(const float4*)(p + 12);
      lo = (v8h){(_Float16)f0.x, (_Float16)f0.y, (_Float16)f0.z, (_Float16)f0.w,
                 (_Float16)f1.x, (_Float16)f1.y, (_Float16)f1.z, (_Float16)f1.w};
      hi = (v8h){(_Float16)f2.x, (_Float16)f2.y, (_Float16)f2.z, (_Float16)f2.w,
                 (_Float16)f3.x, (_Float16)f3.y, (_Float16)f3.z, (_Float16)f3.w};
    }
  };

  // ---- prologue: stage K-tile 0 into buffer 0 ----
#if HAVE_ASYNC_LDS
  if (PREQ) {
    async_cp32((const _Float16*)Ap + aBase, &sA[0][ldsOff]);
    async_cp32((const _Float16*)Bp + bBase, &sB[0][ldsOff]);
    wait_async0();
  } else
#endif
  {
    fetch(aBase, Ap, a0, a1);
    fetch(bBase, Bp, b0, b1);
    _Float16* dA = &sA[0][ldsOff];
    *(v8h*)dA = a0;  *(v8h*)(dA + 8) = a1;
    _Float16* dB = &sB[0][ldsOff];
    *(v8h*)dB = b0;  *(v8h*)(dB + 8) = b1;
  }
  __syncthreads();

  const int KT = K / BK;
  for (int kt = 0; kt < KT; ++kt) {
    const int cur = kt & 1;
    const bool have_next = (kt + 1 < KT);

#if HAVE_ASYNC_LDS
    if (PREQ) {
      if (have_next) {  // DMA next tile straight into the other buffer
        const size_t ko = (size_t)(kt + 1) * BK;
        async_cp32((const _Float16*)Ap + aBase + ko, &sA[cur ^ 1][ldsOff]);
        async_cp32((const _Float16*)Bp + bBase + ko, &sB[cur ^ 1][ldsOff]);
      }
    } else
#endif
    {
      if (have_next) {  // prefetch next tile into registers
        fetch(aBase + (size_t)(kt + 1) * BK, Ap, a0, a1);
        fetch(bBase + (size_t)(kt + 1) * BK, Bp, b0, b1);
      }
    }

    // A fragments: 16-bit A 16x32 layout — lanes 0-15: K{0..7,16..23}, lanes 16-31: K{8..15,24..31}
    v16h af[4];
#pragma unroll
    for (int i = 0; i < 4; ++i) {
      const _Float16* p = &sA[cur][(wm * 64 + i * 16 + lr) * LDT + hid * 8];
      v8h lo = *(const v8h*)p;         // K = hid*8 .. hid*8+7
      v8h hi = *(const v8h*)(p + 16);  // K = 16+hid*8 .. 16+hid*8+7
      af[i] = __builtin_shufflevector(lo, hi, 0, 1, 2, 3, 4, 5, 6, 7,
                                      8, 9, 10, 11, 12, 13, 14, 15);
    }
    // B fragments: 16-bit B 32x16 layout — lanes 0-15: K{0..7,16..23}, lanes 16-31: K{8..15,24..31}
    v16h bf[2];
#pragma unroll
    for (int j = 0; j < 2; ++j) {
      const _Float16* p = &sB[cur][(wn * 32 + j * 16 + lr) * LDT + hid * 8];
      v8h lo = *(const v8h*)p;
      v8h hi = *(const v8h*)(p + 16);
      bf[j] = __builtin_shufflevector(lo, hi, 0, 1, 2, 3, 4, 5, 6, 7,
                                      8, 9, 10, 11, 12, 13, 14, 15);
    }

#pragma unroll
    for (int i = 0; i < 4; ++i)
#pragma unroll
      for (int j = 0; j < 2; ++j)
        acc[i][j] = __builtin_amdgcn_wmma_f32_16x16x32_f16(
            false, af[i], false, bf[j], (short)0, acc[i][j], false, false);

    if (have_next) {
#if HAVE_ASYNC_LDS
      if (PREQ) {
        wait_async0();  // next tile fully landed in LDS
      } else
#endif
      {
        const int nxt = cur ^ 1;
        _Float16* dA = &sA[nxt][ldsOff];
        *(v8h*)dA = a0;  *(v8h*)(dA + 8) = a1;
        _Float16* dB = &sB[nxt][ldsOff];
        *(v8h*)dB = b0;  *(v8h*)(dB + 8) = b1;
      }
      __syncthreads();
    }
  }

  // epilogue: C/D 16x16 f32 layout — VGPR r: row hid*8+r, col = lane%16
#pragma unroll
  for (int j = 0; j < 2; ++j) {
    const int col = bn + wn * 32 + j * 16 + lr;
    const float qb = (float)(_Float16)bias[col];  // inline fp16 quantize of bias
#pragma unroll
    for (int i = 0; i < 4; ++i) {
      const int row0 = bm + wm * 64 + i * 16 + hid * 8;
#pragma unroll
      for (int r = 0; r < 8; ++r)
        C[(size_t)(row0 + r) * N + col] = acc[i][j][r] + qb;
    }
  }
}

extern "C" void kernel_launch(void* const* d_in, const int* in_sizes, int n_in,
                              void* d_out, int out_size, void* d_ws, size_t ws_size,
                              hipStream_t stream) {
  const float* x = (const float*)d_in[0];
  const float* w = (const float*)d_in[1];
  const float* b = (const float*)d_in[2];
  float* out = (float*)d_out;

  const int N = in_sizes[2];            // bias length = OUT
  const int K = in_sizes[1] / N;        // weight is (OUT, IN)
  const int M = in_sizes[0] / K;        // x is (B, IN)

  dim3 grid(N / TILE, M / TILE);
  const size_t needH = ((size_t)M * K + (size_t)N * K) * sizeof(_Float16);

  if (ws_size >= needH) {
    // Pre-quantize x and W to fp16 in workspace: 64MB total -> resident in 192MB L2,
    // so the 32x per-tile re-reads during the GEMM never touch HBM.
    _Float16* Ah = (_Float16*)d_ws;
    _Float16* Bh = Ah + (size_t)M * K;
    int na = M * K;
    qcvt_kernel<<<(na / 4 + 255) / 256, 256, 0, stream>>>(x, Ah, na);
    int nb = N * K;
    qcvt_kernel<<<(nb / 4 + 255) / 256, 256, 0, stream>>>(w, Bh, nb);
    gemm_wmma_kernel<true><<<grid, 256, 0, stream>>>(Ah, Bh, b, out, M, N, K);
  } else {
    // Fallback: load fp32, quantize in-register before LDS store.
    gemm_wmma_kernel<false><<<grid, 256, 0, stream>>>(x, w, b, out, M, N, K);
  }
}
